// BlockLocalSelfAttention_56667798503742
// MI455X (gfx1250) — matrix operand order repaired and checked by reading, not compiled, not run on hardware
//
#include <hip/hip_runtime.h>

// ---------------------------------------------------------------------------
// BigBird-style block-local self-attention for MI455X (gfx1250, wave32, WMMA)
//   n=2, h=16, t=4096, d=64, BLOCK=128, window = 3*128 keys + 1 global token
// Strategy:
//   * one workgroup (8 waves, 256 thr) per (n, h, query-block g)
//   * each wave owns 16 query rows; flash-attention online softmax
//   * S^T = K x Q^T  via v_wmma_f32_16x16x32_f16 (query = lane column)
//   * O^T = V^T x P^T via v_wmma_f32_16x16x32_f16 (P^T built with shfl_xor(16))
//   * K staged [key][d], V staged transposed [d][key] in LDS (f16), 2 phases
// ---------------------------------------------------------------------------

typedef _Float16 h16;
typedef __attribute__((ext_vector_type(16))) _Float16 v16h;
typedef __attribute__((ext_vector_type(8)))  _Float16 h8;
typedef __attribute__((ext_vector_type(4)))  _Float16 h4;
typedef __attribute__((ext_vector_type(8)))  float    v8f;
typedef __attribute__((ext_vector_type(4)))  float    f4;

#define T_SEQ   4096
#define DHEAD   64
#define BLK     128
#define WIN     384      // 3*BLK local window
#define HALFWIN 192      // staged per phase (LDS budget, CU-mode safe)
#define KSTR    72       // K LDS row stride in halves (64 + 8 pad)
#define VSTR    200      // V^T LDS row stride in halves (192 + 8 pad)
#define NEGF    (-1.0e30f)

__device__ __forceinline__ v16h mk16(h8 lo, h8 hi) {
  return __builtin_shufflevector(lo, hi, 0,1,2,3,4,5,6,7,8,9,10,11,12,13,14,15);
}

__global__ __launch_bounds__(256)
void bls_attn_main(const float* __restrict__ Q, const float* __restrict__ K,
                   const float* __restrict__ V, const float* __restrict__ AM,
                   float* __restrict__ O)
{
  __shared__ __align__(16) h16 Klds[HALFWIN * KSTR];   // [key][d]      27.0 KB
  __shared__ __align__(16) h16 Vt[DHEAD * VSTR];       // [d][key]      25.0 KB
  __shared__ float msk[WIN];                           //                1.5 KB
  __shared__ h16 k0h[DHEAD];
  __shared__ h16 v0h[DHEAD];

  const int g   = blockIdx.x;           // query block
  const int hh_ = blockIdx.y;           // head
  const int nb_ = blockIdx.z;           // batch
  const int H   = gridDim.y;
  const int tid = threadIdx.x;
  const long base = (long)(nb_ * H + hh_) * T_SEQ * DHEAD;
  const float* am = AM + (long)nb_ * T_SEQ;
  const float scale = 0.125f;           // 1/sqrt(64)

  // ---- stage mask window (+NEG at pos 0 and out-of-range) and global K0/V0
  for (int w = tid; w < WIN; w += 256) {
    int p = (g - 1) * BLK + w;
    msk[w] = (p > 0 && p < T_SEQ) ? am[p] : NEGF;      // p==0 masked (local)
  }
  if (tid < DHEAD) {
    k0h[tid] = (h16)K[base + tid];
    v0h[tid] = (h16)V[base + tid];
  }

  const int wave = tid >> 5, lane = tid & 31;
  const int m  = lane & 15;             // A-row / D-column index
  const int hf = lane >> 4;             // lane half
  const int qrow = g * BLK + wave * 16 + m;
  const float* qp = Q + base + (long)qrow * DHEAD;

  // ---- Q^T B-fragments, reduce-dim chunks kb = 0, 32 (held in VGPRs)
  // B layout: lane(n=m, half hf) element e <-> k = hf*16 + e
  v16h bq[2];
#pragma unroll
  for (int c = 0; c < 2; ++c) {
    const float* p0 = qp + c * 32 + hf * 16;
    f4 a0 = *(const f4*)(p0 + 0), a1 = *(const f4*)(p0 + 4);
    f4 a2 = *(const f4*)(p0 + 8), a3 = *(const f4*)(p0 + 12);
    h8 lo = {(h16)a0.x,(h16)a0.y,(h16)a0.z,(h16)a0.w,
             (h16)a1.x,(h16)a1.y,(h16)a1.z,(h16)a1.w};
    h8 hi = {(h16)a2.x,(h16)a2.y,(h16)a2.z,(h16)a2.w,
             (h16)a3.x,(h16)a3.y,(h16)a3.z,(h16)a3.w};
    bq[c] = mk16(lo, hi);
  }
  __syncthreads();

  // ---- global-token init of online softmax:  s_g = q . k0 * scale (+0 mask)
  float part = 0.f;
#pragma unroll
  for (int e = 0; e < 16; ++e) {
    part += (float)bq[0][e] * (float)k0h[     hf * 16 + e];
    part += (float)bq[1][e] * (float)k0h[32 + hf * 16 + e];
  }
  float s_g     = (part + __shfl_xor(part, 16)) * scale;
  float run_max = s_g;
  float run_sum = 1.0f;                 // exp(s_g - s_g)
  v8f oacc[4];                          // O^T: dim = t*16 + hf*8 + r, col = query
#pragma unroll
  for (int t = 0; t < 4; ++t)
#pragma unroll
    for (int r = 0; r < 8; ++r) oacc[t][r] = (float)v0h[t * 16 + hf * 8 + r];

  // ---- two staging phases of 192 window keys each
  for (int ph = 0; ph < 2; ++ph) {
    __syncthreads();
    for (int idx = tid; idx < HALFWIN * 16; idx += 256) {
      int kl = idx >> 4, dq = (idx & 15) * 4;
      int p = (g - 1) * BLK + ph * HALFWIN + kl;
      p = (p < 0 || p >= T_SEQ) ? 0 : p;               // clamped rows are masked
      f4 kv = *(const f4*)(K + base + (long)p * DHEAD + dq);
      h4 hk = {(h16)kv.x, (h16)kv.y, (h16)kv.z, (h16)kv.w};
      *(h4*)&Klds[kl * KSTR + dq] = hk;
      f4 vv = *(const f4*)(V + base + (long)p * DHEAD + dq);
      Vt[(dq + 0) * VSTR + kl] = (h16)vv.x;
      Vt[(dq + 1) * VSTR + kl] = (h16)vv.y;
      Vt[(dq + 2) * VSTR + kl] = (h16)vv.z;
      Vt[(dq + 3) * VSTR + kl] = (h16)vv.w;
    }
    __syncthreads();

    for (int js = 0; js < HALFWIN / 32; ++js) {
      const int j0 = js * 32;                 // LDS-local key offset
      const int w0 = ph * HALFWIN + j0;       // window offset (mask index)

      // ---- S^T tiles: d[sub] (sub = key subtile of 16), accumulate over d=64
      v8f d[2];
      d[0] = (v8f){0.f,0.f,0.f,0.f,0.f,0.f,0.f,0.f};
      d[1] = (v8f){0.f,0.f,0.f,0.f,0.f,0.f,0.f,0.f};
#pragma unroll
      for (int sub = 0; sub < 2; ++sub) {
#pragma unroll
        for (int c = 0; c < 2; ++c) {
          // A layout: lane row m, elems: k = c*32 + hf*16 + e
          const h16* kr = &Klds[(j0 + sub * 16 + m) * KSTR + c * 32];
          h8 lo = *(const h8*)(kr + hf * 16);
          h8 hi = *(const h8*)(kr + hf * 16 + 8);
          v16h a = mk16(lo, hi);
          d[sub] = __builtin_amdgcn_wmma_f32_16x16x32_f16(
              false, a, false, bq[c], (short)0, d[sub], false, false);
        }
      }

      // ---- scale + mask; online softmax update (lane = one query column)
      float sv[2][8], tmax = NEGF;
#pragma unroll
      for (int sub = 0; sub < 2; ++sub)
#pragma unroll
        for (int r = 0; r < 8; ++r) {
          float s = d[sub][r] * scale + msk[w0 + sub * 16 + hf * 8 + r];
          sv[sub][r] = s;
          tmax = fmaxf(tmax, s);
        }
      tmax = fmaxf(tmax, __shfl_xor(tmax, 16));        // merge lane halves
      float nm   = fmaxf(run_max, tmax);
      float corr = __expf(run_max - nm);
      run_max = nm;

      float p0[8], p1[8], ts = 0.f;
#pragma unroll
      for (int r = 0; r < 8; ++r) {
        p0[r] = __expf(sv[0][r] - nm);
        p1[r] = __expf(sv[1][r] - nm);
        ts += p0[r] + p1[r];
      }
      ts += __shfl_xor(ts, 16);
      run_sum = run_sum * corr + ts;
#pragma unroll
      for (int t = 0; t < 4; ++t)
#pragma unroll
        for (int r = 0; r < 8; ++r) oacc[t][r] *= corr;

      // ---- build P^T B-fragment: need keys of own subtile from both halves
      v16h bp;
#pragma unroll
      for (int r = 0; r < 8; ++r) {
        float own = hf ? p1[r] : p0[r];                // (subtile hf, key r+8*hf)
        float snd = hf ? p0[r] : p1[r];                // partner needs our other sub
        float rcv = __shfl_xor(snd, 16);               // (subtile hf, key r+8*(1-hf))
        bp[r]     = (h16)(hf ? rcv : own);
        bp[r + 8] = (h16)(hf ? own : rcv);
      }

      // ---- O^T += V^T(16-dim chunk x 32 keys) x P^T(32 keys x 16 queries)
#pragma unroll
      for (int t = 0; t < 4; ++t) {
        const h16* vr = &Vt[(t * 16 + m) * VSTR + j0];
        h8 lo = *(const h8*)(vr + hf * 16);
        h8 hi = *(const h8*)(vr + hf * 16 + 8);
        v16h a = mk16(lo, hi);
        oacc[t] = __builtin_amdgcn_wmma_f32_16x16x32_f16(
            false, a, false, bp, (short)0, oacc[t], false, false);
      }
    }
  }

  // ---- finalize: divide by softmax denominator, store 32B vectors
  float inv = 1.0f / run_sum;
  float* op = O + base + (long)qrow * DHEAD;
#pragma unroll
  for (int t = 0; t < 4; ++t) {
    v8f o = oacc[t];
#pragma unroll
    for (int r = 0; r < 8; ++r) o[r] *= inv;
    *(v8f*)(op + t * 16 + hf * 8) = o;                 // dims t*16+hf*8 .. +7
  }
}

// ---------------------------------------------------------------------------
// Query position 0 gets full attention over all 4096 keys (reference .at[0]).
// Tiny scalar kernel: 32 blocks, runs after the main kernel on the stream.
// ---------------------------------------------------------------------------
__global__ __launch_bounds__(256)
void bls_attn_row0(const float* __restrict__ Q, const float* __restrict__ K,
                   const float* __restrict__ V, const float* __restrict__ AM,
                   float* __restrict__ O, int H)
{
  __shared__ float q0[DHEAD];
  __shared__ float sc[T_SEQ];
  __shared__ float red[8];
  __shared__ float pd[4][DHEAD];
  __shared__ float s_max, s_sum;
  const int bid = blockIdx.x, tid = threadIdx.x;
  const long base = (long)bid * T_SEQ * DHEAD;
  const float* am = AM + (long)(bid / H) * T_SEQ;

  if (tid < DHEAD) q0[tid] = Q[base + tid];
  __syncthreads();

  float lmax = NEGF;
  for (int k = tid; k < T_SEQ; k += 256) {
    const float* kp = K + base + (long)k * DHEAD;
    float s = 0.f;
#pragma unroll 16
    for (int dd = 0; dd < DHEAD; ++dd) s += q0[dd] * kp[dd];
    s = s * 0.125f + am[k];
    sc[k] = s;
    lmax = fmaxf(lmax, s);
  }
  for (int o = 16; o > 0; o >>= 1) lmax = fmaxf(lmax, __shfl_xor(lmax, o));
  if ((tid & 31) == 0) red[tid >> 5] = lmax;
  __syncthreads();
  if (tid == 0) { float mm = red[0]; for (int i = 1; i < 8; ++i) mm = fmaxf(mm, red[i]); s_max = mm; }
  __syncthreads();

  float lsum = 0.f;
  for (int k = tid; k < T_SEQ; k += 256) {
    float p = __expf(sc[k] - s_max);
    sc[k] = p;
    lsum += p;
  }
  for (int o = 16; o > 0; o >>= 1) lsum += __shfl_xor(lsum, o);
  if ((tid & 31) == 0) red[tid >> 5] = lsum;
  __syncthreads();
  if (tid == 0) { float ss = 0.f; for (int i = 0; i < 8; ++i) ss += red[i]; s_sum = ss; }
  __syncthreads();

  const int dd = tid & 63, ch = tid >> 6;
  float acc = 0.f;
  for (int k = ch * (T_SEQ / 4); k < (ch + 1) * (T_SEQ / 4); ++k)
    acc += sc[k] * V[base + (long)k * DHEAD + dd];
  pd[ch][dd] = acc;
  __syncthreads();
  if (ch == 0) {
    float tot = pd[0][dd] + pd[1][dd] + pd[2][dd] + pd[3][dd];
    O[base + dd] = tot / s_sum;
  }
}

extern "C" void kernel_launch(void* const* d_in, const int* in_sizes, int n_in,
                              void* d_out, int out_size, void* d_ws, size_t ws_size,
                              hipStream_t stream)
{
  const float* Q  = (const float*)d_in[0];
  const float* K  = (const float*)d_in[1];
  const float* V  = (const float*)d_in[2];
  const float* AM = (const float*)d_in[3];
  float* O = (float*)d_out;

  const int nT = in_sizes[3];                 // n * t
  const int n  = nT / T_SEQ;
  const int H  = in_sizes[0] / (nT * DHEAD);  // heads

  dim3 grid(T_SEQ / BLK, H, n);
  bls_attn_main<<<grid, 256, 0, stream>>>(Q, K, V, AM, O);
  bls_attn_row0<<<n * H, 256, 0, stream>>>(Q, K, V, AM, O, H);
}
